// TransformerBlock_71605694759791
// MI455X (gfx1250) — compile-verified, matrix operand order checked by reading, not run on hardware
//
#include <hip/hip_runtime.h>
#include <cstdint>

typedef __attribute__((ext_vector_type(16))) _Float16 v16h;
typedef __attribute__((ext_vector_type(8)))  _Float16 v8h;
typedef __attribute__((ext_vector_type(8)))  float    v8f;

constexpr int Bc = 2;            // batch
constexpr int T  = 2048;
constexpr int Dm = 2048;         // model dim
constexpr int H  = 16;
constexpr int HD = 128;
constexpr int Cc = 1024;         // capacity = T/2
constexpr int D3 = 3 * Dm;       // 6144
constexpr float EPS   = 1e-5f;
constexpr float SCALE = 0.08838834764831845f;  // 1/sqrt(128)

// ---------------------------------------------------------------------------
// CDNA5 async global->LDS copy (16B per lane), tracked by ASYNCcnt.
// ---------------------------------------------------------------------------
__device__ __forceinline__ void async_copy_b128(uint32_t lds_byte_off,
                                                const _Float16* g) {
  asm volatile("global_load_async_to_lds_b128 %0, %1, off"
               :: "v"(lds_byte_off), "v"((unsigned long long)(uintptr_t)g)
               : "memory");
}
__device__ __forceinline__ void wait_async_zero() {
  asm volatile("s_wait_asynccnt 0x0" ::: "memory");
}

// ---------------------------------------------------------------------------
// Fragment loader: per CDNA5 ISA 16-bit A layout, lane (r,hl) holds two
// contiguous 8-half chunks at K = hl*8 and K = 16 + hl*8.
// ---------------------------------------------------------------------------
__device__ __forceinline__ v16h load_frag(const _Float16* base, int hl) {
  union { v16h v; v8h h[2]; } u;
  u.h[0] = *(const v8h*)(base + hl * 8);
  u.h[1] = *(const v8h*)(base + 16 + hl * 8);
  return u.v;
}

// ---------------------------------------------------------------------------
// 1) Router logits: one 256-thread block per (b,t) row.
// ---------------------------------------------------------------------------
__global__ __launch_bounds__(256) void router_logits(const float* __restrict__ x,
                                                     const float* __restrict__ wr,
                                                     float* __restrict__ logits) {
  int row = blockIdx.x;  // 0..B*T-1
  const float* xr = x + (size_t)row * Dm;
  float s = 0.f;
  for (int d = threadIdx.x; d < Dm; d += 256) s += xr[d] * wr[d];
  __shared__ float red[256];
  red[threadIdx.x] = s;
  __syncthreads();
  for (int off = 128; off > 0; off >>= 1) {
    if (threadIdx.x < off) red[threadIdx.x] += red[threadIdx.x + off];
    __syncthreads();
  }
  if (threadIdx.x == 0) logits[row] = red[0];
}

// ---------------------------------------------------------------------------
// 2) Top-k (C of T) per batch: bitonic sort desc by (value, lower-index wins),
//    then re-sort the C winners ascending by token index. One block per batch.
// ---------------------------------------------------------------------------
__global__ __launch_bounds__(1024) void topk_sort(const float* __restrict__ logits,
                                                  int* __restrict__ idx,
                                                  float* __restrict__ wsel) {
  __shared__ float sv[T];
  __shared__ int   si[T];
  int b = blockIdx.x, tid = threadIdx.x;
  for (int i = tid; i < T; i += 1024) { sv[i] = logits[b * T + i]; si[i] = i; }
  __syncthreads();
  for (int k = 2; k <= T; k <<= 1) {
    for (int j = k >> 1; j > 0; j >>= 1) {
      for (int i = tid; i < T; i += 1024) {
        int ixj = i ^ j;
        if (ixj > i) {
          float va = sv[i], vb = sv[ixj];
          int ia = si[i], ib = si[ixj];
          bool up = ((i & k) == 0);  // descending sub-sequence
          bool a_ge_b = (va > vb) || (va == vb && ia < ib);
          bool sw = up ? (!a_ge_b) : a_ge_b;
          if (sw) { sv[i] = vb; sv[ixj] = va; si[i] = ib; si[ixj] = ia; }
        }
      }
      __syncthreads();
    }
  }
  // winners occupy [0, C); sort them ascending by token index
  for (int k = 2; k <= Cc; k <<= 1) {
    for (int j = k >> 1; j > 0; j >>= 1) {
      for (int i = tid; i < Cc; i += 1024) {
        int ixj = i ^ j;
        if (ixj > i && ixj < Cc) {
          int ia = si[i], ib = si[ixj];
          bool up = ((i & k) == 0);  // ascending by index
          bool sw = up ? (ia > ib) : (ia < ib);
          if (sw) {
            float tv = sv[i]; sv[i] = sv[ixj]; sv[ixj] = tv;
            si[i] = ib; si[ixj] = ia;
          }
        }
      }
      __syncthreads();
    }
  }
  for (int i = tid; i < Cc; i += 1024) {
    idx[b * Cc + i]  = si[i];
    wsel[b * Cc + i] = sv[i];
  }
}

// ---------------------------------------------------------------------------
// 3) Gather selected rows + RMSNorm + convert to f16. One block per (b,c).
// ---------------------------------------------------------------------------
__global__ __launch_bounds__(256) void gather_rmsnorm_f16(const float* __restrict__ x,
                                                          const int* __restrict__ idx,
                                                          const float* __restrict__ g,
                                                          _Float16* __restrict__ xn) {
  int rc = blockIdx.x;           // 0..B*C-1
  int b = rc / Cc;
  int t = idx[rc];
  const float* xr = x + ((size_t)b * T + t) * Dm;
  float ss = 0.f;
  for (int d = threadIdx.x; d < Dm; d += 256) { float v = xr[d]; ss += v * v; }
  __shared__ float red[256];
  red[threadIdx.x] = ss;
  __syncthreads();
  for (int off = 128; off > 0; off >>= 1) {
    if (threadIdx.x < off) red[threadIdx.x] += red[threadIdx.x + off];
    __syncthreads();
  }
  float rstd = rsqrtf(red[0] / (float)Dm + EPS);
  _Float16* o = xn + (size_t)rc * Dm;
  for (int d = threadIdx.x; d < Dm; d += 256) o[d] = (_Float16)(xr[d] * rstd * g[d]);
}

// ---------------------------------------------------------------------------
// 4) f32 -> f16 weight conversion (grid-stride)
// ---------------------------------------------------------------------------
__global__ __launch_bounds__(256) void cvt_f32_f16(const float* __restrict__ src,
                                                   _Float16* __restrict__ dst, int n) {
  int stride = gridDim.x * blockDim.x;
  for (int i = blockIdx.x * blockDim.x + threadIdx.x; i < n; i += stride)
    dst[i] = (_Float16)src[i];
}

// ---------------------------------------------------------------------------
// 5) QKV GEMM: [2048 x 2048] x [2048 x 6144] -> f16.
//    128x128 block tile, 8 waves (4m x 2n), wave tile 32x64:
//    2 A-frags + 4 B-frags feed 8 v_wmma per k-step.
//    A slab staged via global_load_async_to_lds_b128; B staged transposed.
// ---------------------------------------------------------------------------
__global__ __launch_bounds__(256) void gemm_f16_qkv(const _Float16* __restrict__ A,
                                                    const _Float16* __restrict__ Bm,
                                                    _Float16* __restrict__ Cm) {
  constexpr int K = Dm, N = D3;
  __shared__ __align__(16) _Float16 As[128 * 32];
  __shared__ __align__(16) _Float16 Bt[128 * 32];  // transposed [n][k]
  int row0 = blockIdx.y * 128, col0 = blockIdx.x * 128;
  int tid = threadIdx.x, wid = tid >> 5, lane = tid & 31, hl = lane >> 4, r = lane & 15;
  int wm = wid & 3, wn = wid >> 2;
  uint32_t asbase = (uint32_t)(uintptr_t)&As[0];
  v8f acc[2][4] = {};
  for (int kk = 0; kk < K; kk += 32) {
    {  // async-stage A slab [128 rows x 32 k]
      int arow = tid >> 1, ac = (tid & 1) * 16;
      const _Float16* ga = &A[(size_t)(row0 + arow) * K + kk + ac];
      uint32_t la = asbase + (uint32_t)((arow * 32 + ac) * 2);
      async_copy_b128(la, ga);
      async_copy_b128(la + 16, ga + 8);
    }
    {  // stage B transposed: Bt[n][k]
      int bk = tid >> 3, bn = (tid & 7) * 16;
      const _Float16* gb = &Bm[(size_t)(kk + bk) * N + col0 + bn];
      v8h b0 = *(const v8h*)(gb);
      v8h b1 = *(const v8h*)(gb + 8);
#pragma unroll
      for (int e = 0; e < 8; ++e) {
        Bt[(bn + e) * 32 + bk]     = b0[e];
        Bt[(bn + 8 + e) * 32 + bk] = b1[e];
      }
    }
    if (kk + 32 < K) {  // prefetch next k-slab
      __builtin_prefetch(&A[(size_t)(row0 + (tid >> 1)) * K + kk + 32], 0, 1);
      __builtin_prefetch(&Bm[(size_t)(kk + 32 + (tid >> 3)) * N + col0 + (tid & 7) * 16], 0, 1);
    }
    wait_async_zero();
    __syncthreads();
    v16h af[2], bf[4];
#pragma unroll
    for (int am = 0; am < 2; ++am)
      af[am] = load_frag(&As[(wm * 32 + am * 16 + r) * 32], hl);
#pragma unroll
    for (int nt = 0; nt < 4; ++nt)
      bf[nt] = load_frag(&Bt[(wn * 64 + nt * 16 + r) * 32], hl);
#pragma unroll
    for (int am = 0; am < 2; ++am)
#pragma unroll
      for (int nt = 0; nt < 4; ++nt)
        acc[am][nt] = __builtin_amdgcn_wmma_f32_16x16x32_f16(
            false, af[am], false, bf[nt], (short)0, acc[am][nt], false, false);
    __syncthreads();
  }
#pragma unroll
  for (int am = 0; am < 2; ++am)
#pragma unroll
    for (int nt = 0; nt < 4; ++nt) {
      int cc = col0 + wn * 64 + nt * 16 + r;
#pragma unroll
      for (int i = 0; i < 8; ++i) {
        int rr = row0 + wm * 32 + am * 16 + i + hl * 8;
        Cm[(size_t)rr * N + cc] = (_Float16)acc[am][nt][i];
      }
    }
}

// ---------------------------------------------------------------------------
// 6) Flash attention, causal, per (b, h, 128-row q block). 8 waves; each wave
//    owns a 16-row q strip with full HD=128 accumulator in registers.
//    K tile staged via async global->LDS; V staged transposed manually.
// ---------------------------------------------------------------------------
__global__ __launch_bounds__(256) void flash_attn(const _Float16* __restrict__ qkv,
                                                  _Float16* __restrict__ aout) {
  int qb0 = blockIdx.x * 128;
  int h = blockIdx.y, b = blockIdx.z;
  int tid = threadIdx.x, wid = tid >> 5, lane = tid & 31, hl = lane >> 4, r = lane & 15;
  int qrow0 = qb0 + wid * 16;
  const size_t rstr = (size_t)D3;
  const _Float16* qbase = qkv + (size_t)(b * Cc) * rstr + h * HD;
  const _Float16* kbase = qbase + Dm;
  const _Float16* vbase = qbase + 2 * Dm;

  __shared__ __align__(16) _Float16 Ks[64 * HD];       // [key][hd]
  __shared__ __align__(16) _Float16 Vt[HD * 64];       // [hd][key]
  __shared__ __align__(16) _Float16 Pbuf[8][16][64];   // per-wave P staging
  uint32_t ksbase = (uint32_t)(uintptr_t)&Ks[0];

  v16h qf[4];
#pragma unroll
  for (int kd = 0; kd < 4; ++kd)
    qf[kd] = load_frag(qbase + (size_t)(qrow0 + r) * rstr + kd * 32, hl);

  v8f o[8] = {};
  float mrow[8], lrow[8];
#pragma unroll
  for (int i = 0; i < 8; ++i) { mrow[i] = -1e30f; lrow[i] = 0.f; }

  int jmax = (qb0 + 128) / 64;
  for (int jt = 0; jt < jmax; ++jt) {
    int kt0 = jt * 64;
    {  // stage K [64][128] async, V transposed [128][64] manual
      int krow = tid >> 2, kc = (tid & 3) * 32;
      const _Float16* src = kbase + (size_t)(kt0 + krow) * rstr + kc;
      uint32_t la = ksbase + (uint32_t)((krow * HD + kc) * 2);
#pragma unroll
      for (int c4 = 0; c4 < 4; ++c4)
        async_copy_b128(la + c4 * 16, src + c4 * 8);
      const _Float16* vs = vbase + (size_t)(kt0 + krow) * rstr + kc;
#pragma unroll
      for (int c4 = 0; c4 < 4; ++c4) {
        v8h vv = *(const v8h*)(vs + c4 * 8);
#pragma unroll
        for (int e = 0; e < 8; ++e) Vt[(kc + c4 * 8 + e) * 64 + krow] = vv[e];
      }
    }
    wait_async_zero();
    __syncthreads();
    // S = Q K^T (16 x 64)
    v8f s[4] = {};
#pragma unroll
    for (int nt = 0; nt < 4; ++nt)
#pragma unroll
      for (int kd = 0; kd < 4; ++kd) {
        v16h bf = load_frag(&Ks[(nt * 16 + r) * HD + kd * 32], hl);
        s[nt] = __builtin_amdgcn_wmma_f32_16x16x32_f16(false, qf[kd], false, bf,
                                                       (short)0, s[nt], false, false);
      }
    // causal mask + online softmax (row reductions over 16 lanes of the half)
#pragma unroll
    for (int i = 0; i < 8; ++i) {
      int q = qrow0 + i + hl * 8;
      float sv[4], tmax = -1e30f;
#pragma unroll
      for (int nt = 0; nt < 4; ++nt) {
        int kg = kt0 + nt * 16 + r;
        float v = s[nt][i] * SCALE;
        if (kg > q) v = -1e30f;
        sv[nt] = v;
        tmax = fmaxf(tmax, v);
      }
#pragma unroll
      for (int mk = 1; mk < 16; mk <<= 1) tmax = fmaxf(tmax, __shfl_xor(tmax, mk, 32));
      float mnew = fmaxf(mrow[i], tmax);
      float sc = __expf(mrow[i] - mnew);
      float psum = 0.f;
#pragma unroll
      for (int nt = 0; nt < 4; ++nt) {
        float p = __expf(sv[nt] - mnew);
        Pbuf[wid][i + hl * 8][nt * 16 + r] = (_Float16)p;
        psum += p;
      }
#pragma unroll
      for (int mk = 1; mk < 16; mk <<= 1) psum += __shfl_xor(psum, mk, 32);
      lrow[i] = lrow[i] * sc + psum;
      mrow[i] = mnew;
#pragma unroll
      for (int ot = 0; ot < 8; ++ot) o[ot][i] *= sc;
    }
    // O += P @ V
#pragma unroll
    for (int ks = 0; ks < 2; ++ks) {
      v16h pf = load_frag(&Pbuf[wid][r][ks * 32], hl);
#pragma unroll
      for (int ot = 0; ot < 8; ++ot) {
        v16h vf = load_frag(&Vt[(ot * 16 + r) * 64 + ks * 32], hl);
        o[ot] = __builtin_amdgcn_wmma_f32_16x16x32_f16(false, pf, false, vf,
                                                       (short)0, o[ot], false, false);
      }
    }
    __syncthreads();
  }
#pragma unroll
  for (int ot = 0; ot < 8; ++ot) {
    int hd = ot * 16 + r;
#pragma unroll
    for (int i = 0; i < 8; ++i) {
      int q = qrow0 + i + hl * 8;
      aout[(size_t)(b * Cc + q) * Dm + h * HD + hd] = (_Float16)(o[ot][i] / lrow[i]);
    }
  }
}

// ---------------------------------------------------------------------------
// 7) Projection GEMM (128x128 tiles, async A staging) with fused weighted
//    scatter epilogue: xout[b, idx[c], :] = x + (x + y) * w
//    (tiles partition rows and top-k indices are distinct: no races)
// ---------------------------------------------------------------------------
__global__ __launch_bounds__(256) void gemm_proj_scatter(const _Float16* __restrict__ A,
                                                         const _Float16* __restrict__ Bm,
                                                         const float* __restrict__ x,
                                                         const int* __restrict__ idx,
                                                         const float* __restrict__ wsel,
                                                         float* __restrict__ xout) {
  constexpr int K = Dm, N = Dm;
  __shared__ __align__(16) _Float16 As[128 * 32];
  __shared__ __align__(16) _Float16 Bt[128 * 32];
  int row0 = blockIdx.y * 128, col0 = blockIdx.x * 128;
  int tid = threadIdx.x, wid = tid >> 5, lane = tid & 31, hl = lane >> 4, r = lane & 15;
  int wm = wid & 3, wn = wid >> 2;
  uint32_t asbase = (uint32_t)(uintptr_t)&As[0];
  v8f acc[2][4] = {};
  for (int kk = 0; kk < K; kk += 32) {
    {
      int arow = tid >> 1, ac = (tid & 1) * 16;
      const _Float16* ga = &A[(size_t)(row0 + arow) * K + kk + ac];
      uint32_t la = asbase + (uint32_t)((arow * 32 + ac) * 2);
      async_copy_b128(la, ga);
      async_copy_b128(la + 16, ga + 8);
    }
    {
      int bk = tid >> 3, bn = (tid & 7) * 16;
      const _Float16* gb = &Bm[(size_t)(kk + bk) * N + col0 + bn];
      v8h b0 = *(const v8h*)(gb);
      v8h b1 = *(const v8h*)(gb + 8);
#pragma unroll
      for (int e = 0; e < 8; ++e) {
        Bt[(bn + e) * 32 + bk]     = b0[e];
        Bt[(bn + 8 + e) * 32 + bk] = b1[e];
      }
    }
    if (kk + 32 < K) {
      __builtin_prefetch(&A[(size_t)(row0 + (tid >> 1)) * K + kk + 32], 0, 1);
      __builtin_prefetch(&Bm[(size_t)(kk + 32 + (tid >> 3)) * N + col0 + (tid & 7) * 16], 0, 1);
    }
    wait_async_zero();
    __syncthreads();
    v16h af[2], bf[4];
#pragma unroll
    for (int am = 0; am < 2; ++am)
      af[am] = load_frag(&As[(wm * 32 + am * 16 + r) * 32], hl);
#pragma unroll
    for (int nt = 0; nt < 4; ++nt)
      bf[nt] = load_frag(&Bt[(wn * 64 + nt * 16 + r) * 32], hl);
#pragma unroll
    for (int am = 0; am < 2; ++am)
#pragma unroll
      for (int nt = 0; nt < 4; ++nt)
        acc[am][nt] = __builtin_amdgcn_wmma_f32_16x16x32_f16(
            false, af[am], false, bf[nt], (short)0, acc[am][nt], false, false);
    __syncthreads();
  }
#pragma unroll
  for (int am = 0; am < 2; ++am)
#pragma unroll
    for (int nt = 0; nt < 4; ++nt) {
      int n = col0 + wn * 64 + nt * 16 + r;
#pragma unroll
      for (int i = 0; i < 8; ++i) {
        int rr = row0 + wm * 32 + am * 16 + i + hl * 8;  // rr = b*C + c
        int bb = rr / Cc;
        int trow = idx[rr];
        float wgt = wsel[rr];
        size_t off = ((size_t)bb * T + trow) * Dm + n;
        float xv = x[off];
        xout[off] = xv + (xv + acc[am][nt][i]) * wgt;
      }
    }
}

// ---------------------------------------------------------------------------
// Host-side launch
// ---------------------------------------------------------------------------
extern "C" void kernel_launch(void* const* d_in, const int* in_sizes, int n_in,
                              void* d_out, int out_size, void* d_ws, size_t ws_size,
                              hipStream_t stream) {
  (void)in_sizes; (void)n_in; (void)out_size; (void)ws_size;
  const float* x    = (const float*)d_in[0];
  const float* wr   = (const float*)d_in[1];
  const float* wqkv = (const float*)d_in[2];
  const float* wpr  = (const float*)d_in[3];
  const float* ga   = (const float*)d_in[7];
  // w1/w2/w3/g_ffn (d_in[4..6], d_in[8]) are dead on the routed path.
  float* xout = (float*)d_out;

  char* ws = (char*)d_ws;
  auto alloc = [&](size_t bytes) -> char* {
    char* p = ws;
    ws += (bytes + 255) & ~(size_t)255;
    return p;
  };
  float*    logits = (float*)alloc((size_t)Bc * T * sizeof(float));
  int*      idx    = (int*)alloc((size_t)Bc * Cc * sizeof(int));
  float*    wsel   = (float*)alloc((size_t)Bc * Cc * sizeof(float));
  _Float16* xn     = (_Float16*)alloc((size_t)Bc * Cc * Dm * 2);
  _Float16* wqkv16 = (_Float16*)alloc((size_t)Dm * D3 * 2);
  _Float16* wpr16  = (_Float16*)alloc((size_t)Dm * Dm * 2);
  _Float16* qkv16  = (_Float16*)alloc((size_t)Bc * Cc * D3 * 2);
  _Float16* aout16 = (_Float16*)alloc((size_t)Bc * Cc * Dm * 2);

  hipMemcpyAsync(xout, x, (size_t)Bc * T * Dm * sizeof(float),
                 hipMemcpyDeviceToDevice, stream);
  router_logits<<<Bc * T, 256, 0, stream>>>(x, wr, logits);
  topk_sort<<<Bc, 1024, 0, stream>>>(logits, idx, wsel);
  gather_rmsnorm_f16<<<Bc * Cc, 256, 0, stream>>>(x, idx, ga, xn);
  cvt_f32_f16<<<4096, 256, 0, stream>>>(wqkv, wqkv16, Dm * D3);
  cvt_f32_f16<<<4096, 256, 0, stream>>>(wpr, wpr16, Dm * Dm);
  gemm_f16_qkv<<<dim3(D3 / 128, (Bc * Cc) / 128), 256, 0, stream>>>(xn, wqkv16, qkv16);
  flash_attn<<<dim3(Cc / 128, H, Bc), 256, 0, stream>>>(qkv16, aout16);
  gemm_proj_scatter<<<dim3(Dm / 128, (Bc * Cc) / 128), 256, 0, stream>>>(
      aout16, wpr16, x, idx, wsel, xout);
}
